// MambaEncoder_11433202942005
// MI455X (gfx1250) — matrix operand order repaired and checked by reading, not compiled, not run on hardware
//
#include <hip/hip_runtime.h>
#include <cstdint>

typedef __bf16 bf16;
typedef float  f32x4 __attribute__((ext_vector_type(4)));
typedef float  v8f   __attribute__((ext_vector_type(8)));
typedef bf16   v16bf __attribute__((ext_vector_type(16)));
typedef unsigned int u32x4 __attribute__((ext_vector_type(4)));
typedef unsigned int u32x8 __attribute__((ext_vector_type(8)));

#define B_   8
#define L_   4096
#define DM   512
#define DI   1024
#define NS   16
#define DTR  32
#define XDBW 64   /* DTR + 2*NS */

__device__ __forceinline__ float silu_f(float x) { return x / (1.f + __expf(-x)); }
__device__ __forceinline__ float softplus_f(float x) { return (x > 20.f) ? x : log1pf(__expf(x)); }
__device__ __forceinline__ float act_f(float v, int act) {
  if (act == 1) return silu_f(v);
  if (act == 2) return softplus_f(v);
  return v;
}

// ---------------------------------------------------------------------------
// Main GEMM: out[m,n] = act( sum_k A[m*lda + koff + k] * W[n*ldw + k] + bias[n] )
// A: f32 (converted to bf16 while staging into LDS).
// W: bf16 (pre-converted) -> staged via GLOBAL_LOAD_ASYNC_TO_LDS_B128 (ASYNCcnt).
// Block tile 128(M) x 64(N) x 64(K); 8 waves, each 32x32 via 2x2 WMMA frags,
// 8 v_wmma_f32_16x16x32_bf16 per barrier pair.
// ---------------------------------------------------------------------------
__global__ __launch_bounds__(256)
void gemm_bf16_wmma(const float* __restrict__ A, int lda, int koff,
                    const bf16* __restrict__ W, int ldw,
                    const float* __restrict__ bias,
                    float* __restrict__ outF, bf16* __restrict__ outB,
                    int M, int N, int K, int act) {
  __shared__ alignas(16) bf16 sA[128][72];
  __shared__ alignas(16) bf16 sW[64][72];
  const int lane = threadIdx.x;
  const int wave = threadIdx.y;
  const int tid  = wave * 32 + lane;
  const int m0 = blockIdx.y * 128, n0 = blockIdx.x * 64;
  const int wm = wave >> 1, wn = wave & 1;
  const int l15 = lane & 15, hi = lane >> 4;

  v8f acc[2][2];
#pragma unroll
  for (int i = 0; i < 2; i++)
#pragma unroll
    for (int j = 0; j < 2; j++)
#pragma unroll
      for (int r = 0; r < 8; r++) acc[i][j][r] = 0.f;

  const int arow = tid >> 1;          // 0..127
  const int akb  = (tid & 1) * 32;    // 0 or 32 (floats)
  const int wrow = tid >> 2;          // 0..63
  const int wkb  = (tid & 3) * 16;    // 0,16,32,48 (bf16)

  for (int k0 = 0; k0 < K; k0 += 64) {
    const bool half2 = (k0 + 32 < K);   // second 32-wide K-half valid?

    // Stage A tile (f32 -> bf16): each thread covers one 32-float half-row
    if (akb == 0 || half2) {
      const float* aptr = A + (size_t)(m0 + arow) * lda + koff + k0 + akb;
#pragma unroll
      for (int s = 0; s < 8; ++s) {
        f32x4 v = *(const f32x4*)(aptr + s * 4);
#pragma unroll
        for (int e = 0; e < 4; ++e) sA[arow][akb + s * 4 + e] = (bf16)v[e];
      }
    }
    // Stage W tile: async global->LDS copies, 2 x 16B per thread (ASYNCcnt)
    if (wkb < 32 || half2) {
      const bf16* gw = W + (size_t)(n0 + wrow) * ldw + k0 + wkb;
      unsigned l0 = (unsigned)(uintptr_t)&sW[wrow][wkb];          // LDS offset = addr[31:0]
      unsigned long long g0 = (unsigned long long)(uintptr_t)gw;
      asm volatile("global_load_async_to_lds_b128 %0, %1, off"
                   :: "v"(l0), "v"(g0) : "memory");
      unsigned l1 = (unsigned)(uintptr_t)&sW[wrow][wkb + 8];
      unsigned long long g1 = (unsigned long long)(uintptr_t)(gw + 8);
      asm volatile("global_load_async_to_lds_b128 %0, %1, off"
                   :: "v"(l1), "v"(g1) : "memory");
    }
    asm volatile("s_wait_asynccnt 0x0" ::: "memory");
    __syncthreads();

#pragma unroll
    for (int kk = 0; kk < 2; ++kk) {
      if (kk == 1 && !half2) break;
      const int ko = kk * 32;
      v16bf af[2], bfr[2];
#pragma unroll
      for (int s = 0; s < 2; ++s) {
        // A fragment: lane l15 = row M; e=0..7 -> K=hi*8+e, e=8..15 -> K=16+hi*8+(e-8)
        int m = wm * 32 + s * 16 + l15;
        u32x4 alo = *(const u32x4*)&sA[m][ko + hi * 8];
        u32x4 ahi = *(const u32x4*)&sA[m][ko + 16 + hi * 8];
        u32x8 ta;
#pragma unroll
        for (int q = 0; q < 4; q++) { ta[q] = alo[q]; ta[q + 4] = ahi[q]; }
        af[s] = __builtin_bit_cast(v16bf, ta);
        // B fragment: lane l15 = col N; e=0..7 -> K=hi*8+e, e=8..15 -> K=16+hi*8+(e-8) (B[k][n] = W[n][k])
        int n = wn * 32 + s * 16 + l15;
        u32x4 blo = *(const u32x4*)&sW[n][ko + hi * 8];
        u32x4 bhi = *(const u32x4*)&sW[n][ko + 16 + hi * 8];
        u32x8 tb;
#pragma unroll
        for (int q = 0; q < 4; q++) { tb[q] = blo[q]; tb[q + 4] = bhi[q]; }
        bfr[s] = __builtin_bit_cast(v16bf, tb);
      }
#pragma unroll
      for (int i = 0; i < 2; i++)
#pragma unroll
        for (int j = 0; j < 2; j++)
          acc[i][j] = __builtin_amdgcn_wmma_f32_16x16x32_bf16(
              false, af[i], false, bfr[j], (short)0, acc[i][j], false, false);
    }
    __syncthreads();
  }

  // Epilogue: C layout — VGPR r: row = hi*8 + r, col = l15
#pragma unroll
  for (int i = 0; i < 2; i++) {
#pragma unroll
    for (int j = 0; j < 2; j++) {
      int gn = n0 + wn * 32 + j * 16 + l15;
      float bv = bias ? bias[gn] : 0.f;
      int gmb = m0 + wm * 32 + i * 16 + hi * 8;
#pragma unroll
      for (int r = 0; r < 8; r++) {
        float v = act_f(acc[i][j][r] + bv, act);
        size_t o = (size_t)(gmb + r) * N + gn;
        if (outB) outB[o] = (bf16)v;
        else      outF[o] = v;
      }
    }
  }
}

// ---------------------------------------------------------------------------
// Causal depthwise conv (k=4, left pad 3) + bias + silu.  xa: bf16, xc: f32
// ---------------------------------------------------------------------------
__global__ __launch_bounds__(256)
void conv_silu_kernel(const bf16* __restrict__ xa, const float* __restrict__ cw,
                      const float* __restrict__ cb, float* __restrict__ xc) {
  size_t idx = (size_t)blockIdx.x * 256 + threadIdx.x;   // b*L*DI + t*DI + d
  int d = (int)(idx & (DI - 1));
  size_t bt = idx >> 10;                                  // b*L + t
  int t = (int)(bt & (L_ - 1));
  float acc = cb[d];
#pragma unroll
  for (int j = 0; j < 4; j++) {
    int tt = t - 3 + j;
    if (tt >= 0) acc += cw[d * 4 + j] * (float)xa[(bt - 3 + j) * DI + d];
  }
  xc[idx] = silu_f(acc);
}

// ---------------------------------------------------------------------------
// Selective scan: one thread per (b,d,n) chain, 4096 sequential steps.
// Only the final state h is needed (output taken at last position only).
// ---------------------------------------------------------------------------
__global__ __launch_bounds__(256)
void scan_kernel(const bf16* __restrict__ dt, const float* __restrict__ xc,
                 const float* __restrict__ xdb, const float* __restrict__ An,
                 float* __restrict__ hout) {
  int b = blockIdx.x >> 6;           // 0..7
  int dblk = blockIdx.x & 63;        // 0..63
  int tid = threadIdx.x;
  int d = dblk * 16 + (tid >> 4);
  int n = tid & 15;
  float a = An[d * NS + n];
  const bf16*  dtp = dt  + (size_t)b * L_ * DI + d;
  const float* up  = xc  + (size_t)b * L_ * DI + d;
  const float* Bp  = xdb + (size_t)b * L_ * XDBW + DTR + n;
  float h = 0.f;
  for (int t = 0; t < L_; ++t) {
    float dtv = (float)dtp[(size_t)t * DI];
    float uv  = up[(size_t)t * DI];
    float Bv  = Bp[(size_t)t * XDBW];
    h = __fmaf_rn(h, __expf(dtv * a), dtv * uv * Bv);
    if ((t & 7) == 0 && t + 32 < L_) {
      __builtin_prefetch(up + (size_t)(t + 32) * DI, 0, 1);
      __builtin_prefetch(dtp + (size_t)(t + 32) * DI, 0, 1);
    }
  }
  hout[((size_t)b * DI + d) * NS + n] = h;
}

// ---------------------------------------------------------------------------
// Tail helpers (all O(8 rows), negligible cost)
// ---------------------------------------------------------------------------
__global__ __launch_bounds__(256)
void tail_gemm_kernel(const float* __restrict__ A, long long lda, long long koff,
                      const float* __restrict__ W, const float* __restrict__ bias,
                      float* __restrict__ out, long long ldo, long long ooff,
                      int N, int K, int act) {
  int idx = blockIdx.x * 256 + threadIdx.x;
  int b = idx / N, n = idx - b * N;
  if (b >= B_) return;
  const float* a = A + (long long)b * lda + koff;
  const float* w = W + (long long)n * K;
  float s = bias ? bias[n] : 0.f;
  for (int k = 0; k < K; k++) s = __fmaf_rn(a[k], w[k], s);
  out[(long long)b * ldo + ooff + n] = act_f(s, act);
}

__global__ __launch_bounds__(256)
void tail_convb_kernel(const float* __restrict__ xab, const float* __restrict__ cw,
                       const float* __restrict__ cb, float* __restrict__ xcb) {
  int idx = blockIdx.x * 256 + threadIdx.x;  // 8*1024
  int d = idx & (DI - 1);
  // flipped position 0: only the last conv tap sees real data (rest is padding)
  xcb[idx] = silu_f(cw[d * 4 + 3] * xab[idx] + cb[d]);
}

__global__ __launch_bounds__(256)
void tail_y_kernel(const float* __restrict__ hfin, const float* __restrict__ xdb_f,
                   const float* __restrict__ xc_f, const float* __restrict__ Df,
                   const float* __restrict__ szf, const float* __restrict__ dtb,
                   const float* __restrict__ xcb, const float* __restrict__ xdbb,
                   const float* __restrict__ Db, const float* __restrict__ szb,
                   float* __restrict__ uf, float* __restrict__ ub) {
  int idx = blockIdx.x * 256 + threadIdx.x;   // 8*1024
  int b = idx >> 10, d = idx & (DI - 1);
  // forward: y = h_final . C_last + xc_last * D
  const float* C = xdb_f + ((size_t)(b * L_ + (L_ - 1))) * XDBW + DTR + NS;
  const float* h = hfin + ((size_t)b * DI + d) * NS;
  float y = 0.f;
#pragma unroll
  for (int n = 0; n < NS; n++) y += h[n] * C[n];
  float xcl = xc_f[((size_t)(b * L_ + (L_ - 1))) * DI + d];
  y += xcl * Df[d];
  uf[idx] = y * szf[idx];
  // backward at flipped t=0: h = (dt*u)B, y = dt*u*(B.C) + u*D
  float sBC = 0.f;
#pragma unroll
  for (int n = 0; n < NS; n++)
    sBC += xdbb[b * XDBW + DTR + n] * xdbb[b * XDBW + DTR + NS + n];
  float yb = dtb[idx] * xcb[idx] * sBC + xcb[idx] * Db[d];
  ub[idx] = yb * szb[idx];
}

__global__ __launch_bounds__(512)
void ln_kernel(const float* __restrict__ r, const float* __restrict__ g,
               const float* __restrict__ bb, float* __restrict__ out) {
  __shared__ float red[512];
  int b = blockIdx.x, i = threadIdx.x;
  float v = r[b * 512 + i];
  red[i] = v; __syncthreads();
  for (int s = 256; s > 0; s >>= 1) { if (i < s) red[i] += red[i + s]; __syncthreads(); }
  float mu = red[0] * (1.f / 512.f);
  __syncthreads();
  float dv = v - mu;
  red[i] = dv * dv; __syncthreads();
  for (int s = 256; s > 0; s >>= 1) { if (i < s) red[i] += red[i + s]; __syncthreads(); }
  float var = red[0] * (1.f / 512.f);
  out[b * 512 + i] = dv * rsqrtf(var + 1e-5f) * g[i] + bb[i];
}

__global__ __launch_bounds__(256)
void cvt_bf16_kernel(const float* __restrict__ in, bf16* __restrict__ out, int n) {
  int i = blockIdx.x * 256 + threadIdx.x;
  if (i < n) out[i] = (bf16)in[i];
}

__global__ __launch_bounds__(256)
void negexp_kernel(const float* __restrict__ in, float* __restrict__ out, int n) {
  int i = blockIdx.x * 256 + threadIdx.x;
  if (i < n) out[i] = -__expf(in[i]);
}

// ---------------------------------------------------------------------------
extern "C" void kernel_launch(void* const* d_in, const int* in_sizes, int n_in,
                              void* d_out, int out_size, void* d_ws, size_t ws_size,
                              hipStream_t stream) {
  (void)in_sizes; (void)n_in; (void)out_size; (void)ws_size;
  const float* x         = (const float*)d_in[0];
  const float* proj_w    = (const float*)d_in[1];
  const float* proj_b    = (const float*)d_in[2];
  const float* in_w_f    = (const float*)d_in[3];
  const float* conv_w_f  = (const float*)d_in[4];
  const float* conv_b_f  = (const float*)d_in[5];
  const float* xproj_w_f = (const float*)d_in[6];
  const float* dt_w_f    = (const float*)d_in[7];
  const float* dt_b_f    = (const float*)d_in[8];
  const float* A_log_f   = (const float*)d_in[9];
  const float* D_f       = (const float*)d_in[10];
  const float* out_w_f   = (const float*)d_in[11];
  const float* in_w_b    = (const float*)d_in[12];
  const float* conv_w_b  = (const float*)d_in[13];
  const float* conv_b_b  = (const float*)d_in[14];
  const float* xproj_w_b = (const float*)d_in[15];
  const float* dt_w_b    = (const float*)d_in[16];
  const float* dt_b_b    = (const float*)d_in[17];
  const float* D_b       = (const float*)d_in[19];
  const float* out_w_b   = (const float*)d_in[20];
  const float* fusion_w  = (const float*)d_in[21];
  const float* fusion_b  = (const float*)d_in[22];
  const float* ln_g      = (const float*)d_in[23];
  const float* ln_b      = (const float*)d_in[24];
  float* out = (float*)d_out;

  char* base = (char*)d_ws;
  size_t off = 0;
  auto alloc = [&](size_t bytes) -> char* {
    char* p = base + off;
    off = (off + bytes + 255) & ~(size_t)255;
    return p;
  };
  const size_t ML = (size_t)B_ * L_;              // 32768 rows
  float* xp    = (float*)alloc(ML * DM * 4);      // 64 MB
  bf16*  xa    = (bf16*) alloc(ML * DI * 2);      // 64 MB
  float* xc    = (float*)alloc(ML * DI * 4);      // 128 MB
  float* xdb   = (float*)alloc(ML * XDBW * 4);    // 8 MB
  bf16*  dt    = (bf16*) alloc(ML * DI * 2);      // 64 MB
  float* hfin  = (float*)alloc((size_t)B_ * DI * NS * 4);
  bf16* wbProj = (bf16*)alloc((size_t)DM * DM * 2);
  bf16* wbInA  = (bf16*)alloc((size_t)DI * DM * 2);
  bf16* wbXprj = (bf16*)alloc((size_t)XDBW * DI * 2);
  bf16* wbDtw  = (bf16*)alloc((size_t)DI * DTR * 2);
  float* An    = (float*)alloc((size_t)DI * NS * 4);
  float* t_szf = (float*)alloc(B_ * DI * 4);
  float* t_xab = (float*)alloc(B_ * DI * 4);
  float* t_szb = (float*)alloc(B_ * DI * 4);
  float* t_xcb = (float*)alloc(B_ * DI * 4);
  float* t_xdbb= (float*)alloc(B_ * XDBW * 4);
  float* t_dtb = (float*)alloc(B_ * DI * 4);
  float* t_uf  = (float*)alloc(B_ * DI * 4);
  float* t_ub  = (float*)alloc(B_ * DI * 4);
  float* t_cat = (float*)alloc(B_ * 2 * DM * 4);
  float* t_r   = (float*)alloc(B_ * DM * 4);

  // --- weight prep (bf16 conversion, A = -exp(A_log)) ---
  cvt_bf16_kernel<<<(DM * DM) / 256, 256, 0, stream>>>(proj_w, wbProj, DM * DM);
  cvt_bf16_kernel<<<(DI * DM) / 256, 256, 0, stream>>>(in_w_f, wbInA, DI * DM);
  cvt_bf16_kernel<<<(XDBW * DI) / 256, 256, 0, stream>>>(xproj_w_f, wbXprj, XDBW * DI);
  cvt_bf16_kernel<<<(DI * DTR) / 256, 256, 0, stream>>>(dt_w_f, wbDtw, DI * DTR);
  negexp_kernel<<<(DI * NS) / 256, 256, 0, stream>>>(A_log_f, An, DI * NS);

  dim3 gblk(32, 8);
  // K1: xp = x @ proj_w.T + proj_b              (32768 x 512 x 512)
  gemm_bf16_wmma<<<dim3(DM / 64, ML / 128), gblk, 0, stream>>>(
      x, DM, 0, wbProj, DM, proj_b, xp, nullptr, (int)ML, DM, DM, 0);
  // K2: xa = xp @ in_w_f[0:1024].T   (bf16 out) (32768 x 1024 x 512)
  gemm_bf16_wmma<<<dim3(DI / 64, ML / 128), gblk, 0, stream>>>(
      xp, DM, 0, wbInA, DM, nullptr, nullptr, xa, (int)ML, DI, DM, 0);
  // K3: causal depthwise conv + silu
  conv_silu_kernel<<<(unsigned)(ML * DI / 256), 256, 0, stream>>>(xa, conv_w_f, conv_b_f, xc);
  // K4: xdb = xc @ xproj_w_f.T                  (32768 x 64 x 1024)
  gemm_bf16_wmma<<<dim3(1, ML / 128), gblk, 0, stream>>>(
      xc, DI, 0, wbXprj, DI, nullptr, xdb, nullptr, (int)ML, XDBW, DI, 0);
  // K5: dt = softplus(xdb[:, :32] @ dt_w_f.T + dt_b_f)  (bf16 out)
  gemm_bf16_wmma<<<dim3(DI / 64, ML / 128), gblk, 0, stream>>>(
      xdb, XDBW, 0, wbDtw, DTR, dt_b_f, nullptr, dt, (int)ML, DI, DTR, 2);
  // K6: selective scan -> final state h
  scan_kernel<<<B_ * 64, 256, 0, stream>>>(dt, xc, xdb, An, hfin);

  // --- tail: everything at the last position (fwd) / first flipped position (bwd) ---
  const long long ldaL = (long long)L_ * DM, koffL = (long long)(L_ - 1) * DM;
  tail_gemm_kernel<<<32, 256, 0, stream>>>(xp, ldaL, koffL, in_w_f + DI * DM, nullptr,
                                           t_szf, DI, 0, DI, DM, 1);   // silu(z_f)
  tail_gemm_kernel<<<32, 256, 0, stream>>>(xp, ldaL, koffL, in_w_b, nullptr,
                                           t_xab, DI, 0, DI, DM, 0);   // xa_b
  tail_gemm_kernel<<<32, 256, 0, stream>>>(xp, ldaL, koffL, in_w_b + DI * DM, nullptr,
                                           t_szb, DI, 0, DI, DM, 1);   // silu(z_b)
  tail_convb_kernel<<<32, 256, 0, stream>>>(t_xab, conv_w_b, conv_b_b, t_xcb);
  tail_gemm_kernel<<<2, 256, 0, stream>>>(t_xcb, DI, 0, xproj_w_b, nullptr,
                                          t_xdbb, XDBW, 0, XDBW, DI, 0);
  tail_gemm_kernel<<<32, 256, 0, stream>>>(t_xdbb, XDBW, 0, dt_w_b, dt_b_b,
                                           t_dtb, DI, 0, DI, DTR, 2);
  tail_y_kernel<<<32, 256, 0, stream>>>(hfin, xdb, xc, D_f, t_szf,
                                        t_dtb, t_xcb, t_xdbb, D_b, t_szb, t_uf, t_ub);
  tail_gemm_kernel<<<16, 256, 0, stream>>>(t_uf, DI, 0, out_w_f, nullptr,
                                           t_cat, 2 * DM, 0, DM, DI, 0);
  tail_gemm_kernel<<<16, 256, 0, stream>>>(t_ub, DI, 0, out_w_b, nullptr,
                                           t_cat, 2 * DM, DM, DM, DI, 0);
  tail_gemm_kernel<<<16, 256, 0, stream>>>(t_cat, 2 * DM, 0, fusion_w, fusion_b,
                                           t_r, DM, 0, DM, 2 * DM, 0);
  ln_kernel<<<B_, 512, 0, stream>>>(t_r, ln_g, ln_b, out);
}
